// AC_70506183131618
// MI455X (gfx1250) — matrix operand order repaired and checked by reading, not compiled, not run on hardware
//
#include <hip/hip_runtime.h>
#include <hip/hip_bf16.h>

// ---------------- problem constants ----------------
constexpr int kTN  = 4096;     // nodes
constexpr int kE   = 131072;   // edges
constexpr int kB   = 16;       // graphs
constexpr int kD   = 256;      // model dim
constexpr int kH   = 8;        // heads
constexpr int kC   = 32;       // head dim
constexpr int kDFF = 1024;
constexpr int kL   = 4;        // layers
constexpr int kRB  = 8;
constexpr int kRBE = 128;
constexpr int kINDIM = 96;     // (NPL+NB)*RB
constexpr int kEDIM  = 64;     // PAL*RB*NB

typedef _Float16 v16h __attribute__((ext_vector_type(16)));
typedef float    v8f  __attribute__((ext_vector_type(8)));

// =====================================================================
// Generic WMMA GEMM: out[M,N] = A[M,K] (f32, row major) @ W[N,K]^T + bias
// Wave computes a 16x64 tile (4 sub-tiles share the A fragment).
// Block = 256 threads (8 waves) -> 128x64 tile. M%16==0, N%64==0, K%32==0.
// =====================================================================
template<bool RELU, bool OUT_HALF>
__global__ __launch_bounds__(256)
void wmma_gemm_kernel(const float* __restrict__ A, const float* __restrict__ W,
                      const float* __restrict__ bias, void* __restrict__ outP,
                      int M, int N, int K) {
    const int lane    = threadIdx.x & 31;
    const int wave    = threadIdx.x >> 5;
    const int mTile   = blockIdx.y * 128 + wave * 16;
    if (mTile >= M) return;                       // wave-uniform, EXEC stays all-1s
    const int nBase   = blockIdx.x * 64;
    const int halfSel = lane >> 4;                // 0: lanes 0-15, 1: lanes 16-31
    const int r       = lane & 15;

    const float* aRow = A + (size_t)(mTile + r) * K;
    v8f acc[4] = {};

    for (int k0 = 0; k0 < K; k0 += 32) {
        __builtin_prefetch(aRow + k0 + 32, 0, 1);   // global_prefetch_b8
        // ---- A fragment (16x32 f16 layout, ISA 7.12.2) ----
        v16h a;
        {
            const float* p = aRow + k0 + halfSel * 8;
            #pragma unroll
            for (int j = 0; j < 8; ++j)  a[j]     = (_Float16)p[j];
            #pragma unroll
            for (int j = 0; j < 8; ++j)  a[8 + j] = (_Float16)p[16 + j];
        }
        // ---- 4 B fragments + WMMA ----
        #pragma unroll
        for (int sub = 0; sub < 4; ++sub) {
            const float* wRow = W + (size_t)(nBase + sub * 16 + r) * K + k0 + halfSel * 8;
            v16h b;
            #pragma unroll
            for (int j = 0; j < 16; ++j) b[j] = (_Float16)wRow[(j & 7) + ((j >> 3) << 4)];
            acc[sub] = __builtin_amdgcn_wmma_f32_16x16x32_f16(
                false, a, false, b, (short)0, acc[sub], false, false);
        }
    }
    // ---- store (C/D layout: vgpr i -> row i + 8*halfSel, col = lane&15) ----
    #pragma unroll
    for (int sub = 0; sub < 4; ++sub) {
        const int col = nBase + sub * 16 + r;
        const float bv = bias ? bias[col] : 0.0f;
        #pragma unroll
        for (int i = 0; i < 8; ++i) {
            const int m = mTile + i + halfSel * 8;
            float val = acc[sub][i] + bv;
            if (RELU) val = fmaxf(val, 0.0f);
            if (OUT_HALF) ((_Float16*)outP)[(size_t)m * N + col] = (_Float16)val;
            else          ((float*)outP)[(size_t)m * N + col]    = val;
        }
    }
}

// ---------------- small utility kernels ----------------
__global__ void fill_i32_kernel(int* p, int v, int n) {
    int t = blockIdx.x * blockDim.x + threadIdx.x;
    if (t < n) p[t] = v;
}
__global__ void add_kernel(const float* a, const float* b, float* o, int n) {
    int t = blockIdx.x * blockDim.x + threadIdx.x;
    if (t < n) o[t] = a[t] + b[t];
}
// res = concat(power_alloc[TN,8,8], beam_alloc[TN,8,4], axis=2) -> (TN,96)
__global__ void pack_res_kernel(const float* __restrict__ pa, const float* __restrict__ ba,
                                float* __restrict__ res) {
    int t = blockIdx.x * blockDim.x + threadIdx.x;
    if (t >= kTN * kINDIM) return;
    int n = t / kINDIM, j = t % kINDIM;
    int rb = j / 12, c = j % 12;
    res[t] = (c < 8) ? pa[(n * 8 + rb) * 8 + c] : ba[(n * 8 + rb) * 4 + (c - 8)];
}

// ---------------- CSR build (once per call): edges grouped by dst ----------------
__global__ void deg_kernel(const int* __restrict__ ei, int* __restrict__ deg) {
    int e = blockIdx.x * blockDim.x + threadIdx.x;
    if (e < kE) atomicAdd(&deg[ei[kE + e]], 1);
}
__global__ __launch_bounds__(256)
void scan_kernel(const int* __restrict__ deg, int* __restrict__ rowPtr,
                 int* __restrict__ cursor) {
    // single block: thread 0 does the 4096-wide exclusive scan (once per call)
    if (threadIdx.x == 0) {
        int s = 0;
        for (int i = 0; i < kTN; ++i) { rowPtr[i] = s; s += deg[i]; }
        rowPtr[kTN] = s;
    }
    __syncthreads();
    for (int i = threadIdx.x; i < kTN; i += 256) cursor[i] = rowPtr[i];
}
__global__ void scatter_kernel(const int* __restrict__ ei, int* __restrict__ cursor,
                               int* __restrict__ csr) {
    int e = blockIdx.x * blockDim.x + threadIdx.x;
    if (e >= kE) return;
    int pos = atomicAdd(&cursor[ei[kE + e]], 1);
    csr[pos] = e;
}

// =====================================================================
// Fused per-destination attention with online softmax (flash style).
// Block = 256 threads = 8 waves; wave h handles head h; lane c = channel.
// No atomics, no alpha/amax/den buffers, single pass over incident edges.
// =====================================================================
__global__ __launch_bounds__(256)
void attn_kernel(const float* __restrict__ q, const float* __restrict__ k,
                 const float* __restrict__ v, const _Float16* __restrict__ eh,
                 const int* __restrict__ ei, const int* __restrict__ rowPtr,
                 const int* __restrict__ csr, float* __restrict__ attn) {
    const int dst = blockIdx.x;
    const int h = threadIdx.x >> 5, c = threadIdx.x & 31;
    const int beg = rowPtr[dst], end = rowPtr[dst + 1];
    const float qv = q[(size_t)dst * kD + h * kC + c];

    float m = -3.0e38f, s = 0.0f, acc = 0.0f;
    for (int j = beg; j < end; ++j) {
        const int e   = csr[j];
        const int src = ei[e];
        const size_t eOff = (size_t)e * kD + h * kC + c;
        const size_t sOff = (size_t)src * kD + h * kC + c;
        const float ev = (float)eh[eOff];
        float p = qv * (k[sOff] + ev);
        #pragma unroll
        for (int off = 16; off > 0; off >>= 1) p += __shfl_xor(p, off, 32);
        const float a     = p * 0.17677669529663687f;   // 1/sqrt(32)
        const float mNew  = fmaxf(m, a);
        const float scale = __expf(m - mNew);           // 0 on first edge
        const float pe    = __expf(a - mNew);
        s   = s * scale + pe;
        acc = acc * scale + pe * (v[sOff] + ev);
        m   = mNew;
    }
    attn[(size_t)dst * kD + h * kC + c] = (end > beg) ? acc / s : 0.0f;
}

// ---------------- layernorm: out = LN(a + b [+ c]) * g + beta ----------------
__global__ __launch_bounds__(256)
void ln_kernel(const float* __restrict__ a, const float* __restrict__ b,
               const float* __restrict__ c, const float* __restrict__ g,
               const float* __restrict__ beta, float* __restrict__ out) {
    __shared__ float red[256];
    const int row = blockIdx.x, j = threadIdx.x;
    const size_t idx = (size_t)row * kD + j;
    float v = a[idx] + b[idx] + (c ? c[idx] : 0.0f);
    red[j] = v; __syncthreads();
    for (int s = 128; s > 0; s >>= 1) { if (j < s) red[j] += red[j + s]; __syncthreads(); }
    float mean = red[0] * (1.0f / kD); __syncthreads();
    float d = v - mean;
    red[j] = d * d; __syncthreads();
    for (int s = 128; s > 0; s >>= 1) { if (j < s) red[j] += red[j + s]; __syncthreads(); }
    float var = red[0] * (1.0f / kD);
    out[idx] = d * rsqrtf(var + 1e-5f) * g[j] + beta[j];
}

// ---------------- heads ----------------
__global__ __launch_bounds__(256)
void mean_kernel(const float* __restrict__ x, float* __restrict__ mean) {
    const int b = blockIdx.x, j = threadIdx.x;
    float s = 0.0f;
    for (int i = 0; i < kTN / kB; ++i) s += x[((size_t)b * (kTN / kB) + i) * kD + j];
    mean[b * kD + j] = s * (float)kB / (float)kTN;
}
__global__ void value_kernel(const float* __restrict__ mean, const float* __restrict__ Wc,
                             const float* __restrict__ bc, float* __restrict__ value) {
    int b = threadIdx.x; if (b >= kB) return;
    float s = bc[0];
    for (int j = 0; j < kD; ++j) s += mean[b * kD + j] * Wc[j];
    value[b] = s;
}
__global__ void actor_kernel(const float* __restrict__ x, const float* __restrict__ Wa,
                             const float* __restrict__ ba, float* __restrict__ logits) {
    int t = blockIdx.x * blockDim.x + threadIdx.x;
    if (t >= kTN * kRB) return;
    int node = t >> 3, r = t & 7;
    const float* xr = x + (size_t)node * kD;
    const float* wr = Wa + r * kD;
    float s = ba[r];
    for (int j = 0; j < kD; ++j) s += xr[j] * wr[j];
    logits[t] = s;
}

__device__ inline unsigned long long splitmix64(unsigned long long z) {
    z += 0x9E3779B97F4A7C15ull;
    z = (z ^ (z >> 30)) * 0xBF58476D1CE4E5B9ull;
    z = (z ^ (z >> 27)) * 0x94D049BB133111EBull;
    return z ^ (z >> 31);
}
__device__ inline float uniform01(unsigned long long s) {
    return ((float)(splitmix64(s) >> 40) + 0.5f) * (1.0f / 16777216.0f);
}

__global__ __launch_bounds__(256)
void hi_sample_kernel(const float* __restrict__ logits, int* __restrict__ link_out,
                      int* __restrict__ rb_out, float* __restrict__ lr_lp,
                      float* __restrict__ hi_ent) {
    __shared__ float red[256];
    __shared__ float s_max, s_sum;
    const int b = blockIdx.x, t = threadIdx.x;
    const int NL = (kTN / kB) * kRB;              // 2048
    const float* L = logits + (size_t)b * NL;
    float mx = -3.4e38f;
    for (int j = t; j < NL; j += 256) mx = fmaxf(mx, L[j]);
    red[t] = mx; __syncthreads();
    for (int s = 128; s > 0; s >>= 1) { if (t < s) red[t] = fmaxf(red[t], red[t + s]); __syncthreads(); }
    if (t == 0) s_max = red[0]; __syncthreads();
    const float mxv = s_max;
    float sum = 0.0f;
    for (int j = t; j < NL; j += 256) sum += __expf(L[j] - mxv);
    red[t] = sum; __syncthreads();
    for (int s = 128; s > 0; s >>= 1) { if (t < s) red[t] += red[t + s]; __syncthreads(); }
    if (t == 0) s_sum = red[0]; __syncthreads();
    const float lse = mxv + __logf(s_sum);
    float ent = 0.0f;
    for (int j = t; j < NL; j += 256) { float lp = L[j] - lse; ent -= __expf(lp) * lp; }
    red[t] = ent; __syncthreads();
    for (int s = 128; s > 0; s >>= 1) { if (t < s) red[t] += red[t + s]; __syncthreads(); }
    if (t == 0) {
        hi_ent[b] = red[0];
        float target = uniform01(0x2a000000ull + (unsigned)b) * s_sum;
        float csum = 0.0f; int pick = NL - 1;
        for (int j = 0; j < NL; ++j) {
            csum += __expf(L[j] - mxv);
            if (csum >= target) { pick = j; break; }
        }
        link_out[b] = pick >> 3;
        rb_out[b]   = pick & 7;
        lr_lp[b]    = L[pick] - lse;
    }
}

__global__ void bi_kernel(const float* __restrict__ x, const float* __restrict__ rb_table,
                          const int* __restrict__ ptr, const int* __restrict__ link,
                          const int* __restrict__ rb, float* __restrict__ bi) {
    const int b = blockIdx.x, j = threadIdx.x;      // 384 threads
    const int node = ptr[b] + link[b];
    float v = (j < kD) ? x[(size_t)node * kD + j]
                       : rb_table[rb[b] * kRBE + (j - kD)];
    bi[b * (kD + kRBE) + j] = v;
}
__global__ void blog_kernel(const float* __restrict__ mid, const float* __restrict__ Wm2,
                            const float* __restrict__ bm2, float* __restrict__ blog) {
    int t = blockIdx.x * blockDim.x + threadIdx.x;
    if (t >= kB * 4) return;
    int b = t >> 2, n = t & 3;
    const float* mr = mid + b * kD;
    const float* wr = Wm2 + n * kD;
    float s = bm2[n];
    for (int j = 0; j < kD; ++j) s += mr[j] * wr[j];
    blog[t] = s;
}
__global__ void lo_sample_kernel(const float* __restrict__ blog, int* __restrict__ beam,
                                 float* __restrict__ b_lp, float* __restrict__ lo_ent) {
    int b = threadIdx.x; if (b >= kB) return;
    const float* L = blog + b * 4;
    float mx = fmaxf(fmaxf(L[0], L[1]), fmaxf(L[2], L[3]));
    float ex[4], sum = 0.0f;
    for (int j = 0; j < 4; ++j) { ex[j] = __expf(L[j] - mx); sum += ex[j]; }
    float lse = mx + __logf(sum);
    float ent = 0.0f;
    for (int j = 0; j < 4; ++j) { float lp = L[j] - lse; ent -= __expf(lp) * lp; }
    float target = uniform01(0x2b000000ull + (unsigned)b) * sum;
    float csum = 0.0f; int pick = 3;
    for (int j = 0; j < 4; ++j) { csum += ex[j]; if (csum >= target) { pick = j; break; } }
    beam[b] = pick; b_lp[b] = L[pick] - lse; lo_ent[b] = ent;
}
__global__ void finalize_kernel(const int* __restrict__ link, const int* __restrict__ rb,
                                const int* __restrict__ beam, const float* __restrict__ lr_lp,
                                const float* __restrict__ b_lp, const float* __restrict__ hi_ent,
                                const float* __restrict__ lo_ent, const float* __restrict__ value,
                                float* __restrict__ out) {
    int t = threadIdx.x;                  // 128 threads
    if (t < 48) {
        int b = t / 3, c = t % 3;
        int v = (c == 0) ? link[b] : (c == 1) ? rb[b] : beam[b];
        out[t] = (float)v;
    } else if (t < 64)  out[t] = lr_lp[t - 48] + b_lp[t - 48];
    else if (t < 80)    out[t] = hi_ent[t - 64] + lo_ent[t - 64];
    else if (t < 96)    out[t] = value[t - 80];
    else if (t < 112)   out[t] = hi_ent[t - 96];
    else if (t < 128)   out[t] = lo_ent[t - 112];
}

// =====================================================================
extern "C" void kernel_launch(void* const* d_in, const int* in_sizes, int n_in,
                              void* d_out, int out_size, void* d_ws, size_t ws_size,
                              hipStream_t stream) {
    (void)in_sizes; (void)n_in; (void)out_size; (void)ws_size;
    // -------- inputs (jax tree_util alphabetical flatten of params) --------
    const float* power_alloc = (const float*)d_in[0];
    const float* beam_alloc  = (const float*)d_in[1];
    const float* npa         = (const float*)d_in[2];   // (TN,64)
    const float* epa         = (const float*)d_in[3];   // (E,64)
    const int*   ei          = (const int*)d_in[4];     // (2,E)
    const int*   ptr         = (const int*)d_in[5];
    // d_in[6] = batch (unused: contiguous blocks of 256 nodes per graph)
    const float* W_actor  = (const float*)d_in[7];
    const float* W_critic = (const float*)d_in[8];
    const float* W_emb    = (const float*)d_in[9];
    const float* W_in     = (const float*)d_in[10];
    const float* Wm1      = (const float*)d_in[11];
    const float* Wm2      = (const float*)d_in[12];
    const float* b_actor  = (const float*)d_in[13];
    const float* b_critic = (const float*)d_in[14];
    const float* b_emb    = (const float*)d_in[15];
    const float* b_in     = (const float*)d_in[16];
    const float* bm1      = (const float*)d_in[17];
    const float* bm2      = (const float*)d_in[18];
    // layers: 4 x [W1,W2,We,Wk,Wq,Wskip,Wv,b1,b2,bk,bq,bskip,bv,n1b,n1g,n2b,n2g]
    const int LAYER0 = 19, LPN = 17;
    const float* rb_table = (const float*)d_in[LAYER0 + kL * LPN];

    // -------- workspace carve --------
    char* wsb = (char*)d_ws;
    size_t off = 0;
    auto carve = [&](size_t bytes) -> char* {
        char* p = wsb + off;
        off = (off + bytes + 255) & ~(size_t)255;
        return p;
    };
    float*    res    = (float*)carve((size_t)kTN * kINDIM * 4);
    float*    inp    = (float*)carve((size_t)kTN * kD * 4);
    float*    x      = (float*)carve((size_t)kTN * kD * 4);
    float*    xin    = (float*)carve((size_t)kTN * kD * 4);
    float*    q      = (float*)carve((size_t)kTN * kD * 4);
    float*    k      = (float*)carve((size_t)kTN * kD * 4);
    float*    v      = (float*)carve((size_t)kTN * kD * 4);
    float*    sk     = (float*)carve((size_t)kTN * kD * 4);
    float*    attn   = (float*)carve((size_t)kTN * kD * 4);
    float*    x1     = (float*)carve((size_t)kTN * kD * 4);
    float*    h1     = (float*)carve((size_t)kTN * kDFF * 4);
    float*    h2     = (float*)carve((size_t)kTN * kD * 4);
    _Float16* eh     = (_Float16*)carve((size_t)kE * kD * 2);
    int*      deg    = (int*)carve((size_t)kTN * 4);
    int*      cursor = (int*)carve((size_t)kTN * 4);
    int*      rowPtr = (int*)carve((size_t)(kTN + 1) * 4);
    int*      csr    = (int*)carve((size_t)kE * 4);
    float*    meanb  = (float*)carve((size_t)kB * kD * 4);
    float*    logit  = (float*)carve((size_t)kTN * kRB * 4);
    float*    valb   = (float*)carve((size_t)kB * 4);
    int*      linkb  = (int*)carve((size_t)kB * 4);
    int*      rbb    = (int*)carve((size_t)kB * 4);
    int*      beamb  = (int*)carve((size_t)kB * 4);
    float*    lrlp   = (float*)carve((size_t)kB * 4);
    float*    hient  = (float*)carve((size_t)kB * 4);
    float*    blp    = (float*)carve((size_t)kB * 4);
    float*    loent  = (float*)carve((size_t)kB * 4);
    float*    bi     = (float*)carve((size_t)kB * (kD + kRBE) * 4);
    float*    mid    = (float*)carve((size_t)kB * kD * 4);
    float*    blog   = (float*)carve((size_t)kB * 4 * 4);

    auto gemm = [&](const float* A, const float* W_, const float* bias, void* out,
                    int M, int N, int K, bool relu, bool outHalf) {
        dim3 g(N / 64, (M + 127) / 128), blk(256);
        if (outHalf)
            wmma_gemm_kernel<false, true><<<g, blk, 0, stream>>>(A, W_, bias, out, M, N, K);
        else if (relu)
            wmma_gemm_kernel<true, false><<<g, blk, 0, stream>>>(A, W_, bias, out, M, N, K);
        else
            wmma_gemm_kernel<false, false><<<g, blk, 0, stream>>>(A, W_, bias, out, M, N, K);
    };

    // -------- input projections --------
    pack_res_kernel<<<(kTN * kINDIM + 255) / 256, 256, 0, stream>>>(power_alloc, beam_alloc, res);
    gemm(res, W_in,  b_in,  inp, kTN, kD, kINDIM, false, false);   // inp
    gemm(npa, W_emb, b_emb, x,   kTN, kD, kEDIM,  false, false);   // x0

    // -------- CSR by destination (once; edge_index is call-invariant) --------
    fill_i32_kernel<<<(kTN + 255) / 256, 256, 0, stream>>>(deg, 0, kTN);
    deg_kernel<<<(kE + 255) / 256, 256, 0, stream>>>(ei, deg);
    scan_kernel<<<1, 256, 0, stream>>>(deg, rowPtr, cursor);
    scatter_kernel<<<(kE + 255) / 256, 256, 0, stream>>>(ei, cursor, csr);

    // -------- transformer layers --------
    const int nND = kTN * kD;
    for (int l = 0; l < kL; ++l) {
        const int base = LAYER0 + l * LPN;
        const float* W1    = (const float*)d_in[base + 0];
        const float* W2    = (const float*)d_in[base + 1];
        const float* We    = (const float*)d_in[base + 2];
        const float* Wk    = (const float*)d_in[base + 3];
        const float* Wq    = (const float*)d_in[base + 4];
        const float* Wskip = (const float*)d_in[base + 5];
        const float* Wv    = (const float*)d_in[base + 6];
        const float* b1    = (const float*)d_in[base + 7];
        const float* b2    = (const float*)d_in[base + 8];
        const float* bk    = (const float*)d_in[base + 9];
        const float* bq    = (const float*)d_in[base + 10];
        const float* bskip = (const float*)d_in[base + 11];
        const float* bv    = (const float*)d_in[base + 12];
        const float* n1b   = (const float*)d_in[base + 13];
        const float* n1g   = (const float*)d_in[base + 14];
        const float* n2b   = (const float*)d_in[base + 15];
        const float* n2g   = (const float*)d_in[base + 16];

        add_kernel<<<(nND + 255) / 256, 256, 0, stream>>>(x, inp, xin, nND);

        gemm(xin, Wq,    bq,    q,  kTN, kD, kD, false, false);
        gemm(xin, Wk,    bk,    k,  kTN, kD, kD, false, false);
        gemm(xin, Wv,    bv,    v,  kTN, kD, kD, false, false);
        gemm(xin, Wskip, bskip, sk, kTN, kD, kD, false, false);
        gemm(epa, We, nullptr, eh, kE, kD, kEDIM, false, true);   // edge embed -> f16

        attn_kernel<<<kTN, 256, 0, stream>>>(q, k, v, eh, ei, rowPtr, csr, attn);

        ln_kernel<<<kTN, 256, 0, stream>>>(xin, attn, sk, n1g, n1b, x1);   // x1 = LN(xin+out)
        gemm(x1, W1, b1, h1, kTN, kDFF, kD, true, false);                  // relu FFN
        gemm(h1, W2, b2, h2, kTN, kD, kDFF, false, false);
        ln_kernel<<<kTN, 256, 0, stream>>>(x1, h2, nullptr, n2g, n2b, x);  // x = LN(x1+h)
    }

    // -------- heads --------
    mean_kernel<<<kB, 256, 0, stream>>>(x, meanb);
    value_kernel<<<1, 32, 0, stream>>>(meanb, W_critic, b_critic, valb);
    actor_kernel<<<(kTN * kRB + 255) / 256, 256, 0, stream>>>(x, W_actor, b_actor, logit);
    hi_sample_kernel<<<kB, 256, 0, stream>>>(logit, linkb, rbb, lrlp, hient);
    bi_kernel<<<kB, kD + kRBE, 0, stream>>>(x, rb_table, ptr, linkb, rbb, bi);
    gemm(bi, Wm1, bm1, mid, kB, kD, kD + kRBE, true, false);   // 16x256x384 WMMA + relu
    blog_kernel<<<1, 64, 0, stream>>>(mid, Wm2, bm2, blog);
    lo_sample_kernel<<<1, 32, 0, stream>>>(blog, beamb, blp, loent);
    finalize_kernel<<<1, 128, 0, stream>>>(linkb, rbb, beamb, lrlp, blp, hient, loent,
                                           valb, (float*)d_out);
}
